// ProperTRM_80590766342549
// MI455X (gfx1250) — matrix operand order repaired and checked by reading, not compiled, not run on hardware
//
#include <hip/hip_runtime.h>

#define NB   262144
#define DIN  256
#define HID  32
#define EXPD 64
#define NWV  4            // waves per block (wave32)

typedef __attribute__((ext_vector_type(16))) _Float16 v16h;
typedef __attribute__((ext_vector_type(8)))  float    v8f;

union AF { v16h h; unsigned u[8]; };
union VF { v8f v; float f[8]; };

__device__ __forceinline__ void lds_fence() {
  // within-wave LDS ordering + compiler barrier for cross-lane LDS traffic
  asm volatile("s_wait_dscnt 0x0" ::: "memory");
}

template <int N>
__device__ __forceinline__ void wait_async() {
  asm volatile("s_wait_asynccnt %0" :: "i"(N) : "memory");
}

// Async DMA: 16 bytes global -> LDS, tracked by ASYNCcnt (no VGPR data path).
__device__ __forceinline__ void async_ld_b128(unsigned lds_off, const void* g) {
  asm volatile("global_load_async_to_lds_b128 %0, %1, off"
               :: "v"(lds_off), "v"((unsigned long long)(uintptr_t)g)
               : "memory");
}

__device__ __forceinline__ v8f wmma_f16(const AF a, const AF b, v8f c) {
  return __builtin_amdgcn_wmma_f32_16x16x32_f16(
      false, a.h, false, b.h, (short)0, c, false, false);
}

__device__ __forceinline__ unsigned f16pk(float a, float b) {
  union { _Float16 h[2]; unsigned u; } t;
  t.h[0] = (_Float16)a; t.h[1] = (_Float16)b;
  return t.u;
}

// A-fragment K index for 16-bit A 16x32 (ISA 7.12.2): VGPR v, lane-half h
__device__ __forceinline__ int afrag_k(int v, int half) {
  return ((v & 3) << 1) + ((v >> 2) << 4) + half * 8;
}

struct __align__(16) Smem {
  unsigned pinw[16 * 32 * 8];        // 16 KB  in_w as packed B-fragments
  unsigned pw1 [4 * 32 * 8];         // 4 KB   w1 fragments
  unsigned pw2 [4 * 32 * 8];         // 4 KB   w2 fragments
  unsigned pw3 [4 * 32 * 8];         // 4 KB   w3 fragments
  float    xsf[NWV][2][16][32];      // 16 KB  double-buffered async x stage (f32)
  _Float16 zz [NWV][16][HID + 2];    // padded: row stride 17 dwords (conflict-free)
  _Float16 act[NWV][16][EXPD + 2];   // padded: row stride 33 dwords
  float    lg [NWV][16][2];
  float inb[HID], nrm[HID], hin[HID], lin[HID];
  float ow[HID][2], ob[2];
};

// Packed-fragment load: [frag][lane][v] dwords -> two ds_load_b128 per fragment
__device__ __forceinline__ void load_pfrag(AF& b, const unsigned* base, int frag, int lane) {
  const uint4* p = (const uint4*)(base + ((frag * 32 + lane) << 3));
  uint4 q0 = p[0], q1 = p[1];
  b.u[0] = q0.x; b.u[1] = q0.y; b.u[2] = q0.z; b.u[3] = q0.w;
  b.u[4] = q1.x; b.u[5] = q1.y; b.u[6] = q1.z; b.u[7] = q1.w;
}

// One SwiGLU block (z -> rmsnorm(z + mlp(z)) * g) on a 16x32 tile held as
// two f32 D-fragments (z0 = cols 0..15, z1 = cols 16..31).
__device__ __forceinline__ void block16(VF& z0, VF& z1,
    _Float16 (*zz)[HID + 2], _Float16 (*act)[EXPD + 2],
    const AF bw1[4], const AF bw2[4], const AF (*bw3)[2],
    float g0, float g1, int lane)
{
  const int col = lane & 15, half = lane >> 4;
#pragma unroll
  for (int r = 0; r < 8; ++r) {
    zz[r + 8 * half][col]      = (_Float16)z0.f[r];
    zz[r + 8 * half][col + 16] = (_Float16)z1.f[r];
  }
  lds_fence();
  AF az;                                  // A-fragment of z (16x32)
#pragma unroll
  for (int v = 0; v < 8; ++v)
    az.u[v] = *(const unsigned*)&zz[col][afrag_k(v, half)];
  lds_fence();
#pragma unroll
  for (int n = 0; n < 4; ++n) {
    v8f zero = {};
    VF cu, cv;
    cu.v = wmma_f16(az, bw1[n], zero);
    cv.v = wmma_f16(az, bw2[n], zero);
#pragma unroll
    for (int r = 0; r < 8; ++r) {
      float u = cu.f[r];
      float s = u / (1.f + __expf(-u));   // silu
      act[r + 8 * half][n * 16 + col] = (_Float16)(s * cv.f[r]);
    }
  }
  lds_fence();
  AF a0, a1;                              // A-fragments of gate (16x64)
#pragma unroll
  for (int v = 0; v < 8; ++v) {
    int k = afrag_k(v, half);
    a0.u[v] = *(const unsigned*)&act[col][k];
    a1.u[v] = *(const unsigned*)&act[col][32 + k];
  }
  lds_fence();
  v8f h0 = {}, h1 = {};
  h0 = wmma_f16(a0, bw3[0][0], h0);
  h0 = wmma_f16(a1, bw3[1][0], h0);
  h1 = wmma_f16(a0, bw3[0][1], h1);
  h1 = wmma_f16(a1, bw3[1][1], h1);
  VF y0, y1; y0.v = h0; y1.v = h1;
#pragma unroll
  for (int r = 0; r < 8; ++r) { y0.f[r] += z0.f[r]; y1.f[r] += z1.f[r]; }
#pragma unroll
  for (int r = 0; r < 8; ++r) {           // RMSNorm over 32 cols per row
    float s = y0.f[r] * y0.f[r] + y1.f[r] * y1.f[r];
    s += __shfl_xor(s, 1, 32);
    s += __shfl_xor(s, 2, 32);
    s += __shfl_xor(s, 4, 32);
    s += __shfl_xor(s, 8, 32);
    float sc = rsqrtf(s * (1.f / 32.f) + 1e-6f);
    z0.f[r] = y0.f[r] * sc * g0;
    z1.f[r] = y1.f[r] * sc * g1;
  }
}

// issue one 16x32 f32 chunk of x into an LDS stage buffer via async DMA
__device__ __forceinline__ void issue_chunk(const float* x, long long row0, int ch,
                                            float (*dst)[32], int lane) {
#pragma unroll
  for (int t = 0; t < 4; ++t) {
    int idx = t * 32 + lane;             // 128 x b128 = 16 rows x 32 f32
    int r = idx >> 3, c4 = idx & 7;
    async_ld_b128((unsigned)(uintptr_t)&dst[r][c4 * 4],
                  x + (row0 + r) * DIN + ch * 32 + c4 * 4);
  }
}

__global__ __launch_bounds__(128) void trm_fused(
    const float* __restrict__ x,   const float* __restrict__ in_w,
    const float* __restrict__ in_b, const float* __restrict__ w1,
    const float* __restrict__ w2,  const float* __restrict__ w3,
    const float* __restrict__ norm_w, const float* __restrict__ Hin,
    const float* __restrict__ Lin, const float* __restrict__ ow,
    const float* __restrict__ ob,  float* __restrict__ out)
{
  __shared__ Smem sh;
  const int tid = threadIdx.x, wv = tid >> 5, lane = tid & 31;
  const int col = lane & 15, half = lane >> 4;

  // ---- cooperative init: pack all weights directly into B-fragment order
  // layout: dword index = frag*256 + lane*8 + v ; each dword = f16{w[k][c], w[k+1][c]}
  for (int i = tid; i < 4 * 32 * 8; i += 128) {       // pw1 / pw2 (K=32, 4 N-tiles)
    int ln = (i >> 3) & 31, v = i & 7, n = i >> 8;
    int k = ((ln >> 4) << 3) + ((v & 3) << 1) + ((v >> 2) << 4);
    int c = n * 16 + (ln & 15);
    sh.pw1[i] = f16pk(w1[k * EXPD + c], w1[(k + 1) * EXPD + c]);
    sh.pw2[i] = f16pk(w2[k * EXPD + c], w2[(k + 1) * EXPD + c]);
  }
  for (int i = tid; i < 4 * 32 * 8; i += 128) {       // pw3 (2 K-blocks x 2 N-tiles)
    int ln = (i >> 3) & 31, v = i & 7, fr = i >> 8;
    int kb = fr >> 1, n = fr & 1;
    int k = kb * 32 + ((ln >> 4) << 3) + ((v & 3) << 1) + ((v >> 2) << 4);
    int c = n * 16 + (ln & 15);
    sh.pw3[i] = f16pk(w3[k * HID + c], w3[(k + 1) * HID + c]);
  }
  for (int i = tid; i < 16 * 32 * 8; i += 128) {      // pinw (8 K-blocks x 2 N-tiles)
    int ln = (i >> 3) & 31, v = i & 7, fr = i >> 8;
    int ch = fr >> 1, n = fr & 1;
    int k = ch * 32 + ((ln >> 4) << 3) + ((v & 3) << 1) + ((v >> 2) << 4);
    int c = n * 16 + (ln & 15);
    sh.pinw[i] = f16pk(in_w[k * HID + c], in_w[(k + 1) * HID + c]);
  }
  if (tid < HID) {
    sh.inb[tid] = in_b[tid]; sh.nrm[tid] = norm_w[tid];
    sh.hin[tid] = Hin[tid];  sh.lin[tid] = Lin[tid];
    sh.ow[tid][0] = ow[2 * tid]; sh.ow[tid][1] = ow[2 * tid + 1];
  }
  if (tid < 2) sh.ob[tid] = ob[tid];
  __syncthreads();

  // ---- persistent per-wave B-fragments (two ds_load_b128 each)
  AF bw1f[4], bw2f[4], bw3f[2][2];
#pragma unroll
  for (int n = 0; n < 4; ++n) {
    load_pfrag(bw1f[n], sh.pw1, n, lane);
    load_pfrag(bw2f[n], sh.pw2, n, lane);
  }
#pragma unroll
  for (int kb = 0; kb < 2; ++kb)
#pragma unroll
    for (int n = 0; n < 2; ++n)
      load_pfrag(bw3f[kb][n], sh.pw3, kb * 2 + n, lane);

  const long long row0 = ((long long)blockIdx.x * NWV + wv) * 16;

  // ---- x_proj = x @ in_w + in_b : async double-buffered 32-col chunks
  v8f acc0 = {}, acc1 = {};
  issue_chunk(x, row0, 0, sh.xsf[wv][0], lane);
#pragma unroll
  for (int ch = 0; ch < 8; ++ch) {
    if (ch < 7) {
      issue_chunk(x, row0, ch + 1, sh.xsf[wv][(ch + 1) & 1], lane);
      wait_async<4>();      // chunk ch complete; ch+1 (4 xfers) still in flight
    } else {
      wait_async<0>();
    }
    AF ax;                  // A-fragment from f32 stage: ds_load_b64 + cvt_pk
#pragma unroll
    for (int v = 0; v < 8; ++v) {
      int k = afrag_k(v, half);
      float2 d = *(const float2*)&sh.xsf[wv][ch & 1][col][k];
      ax.u[v] = f16pk(d.x, d.y);
    }
    AF b0, b1;
    load_pfrag(b0, sh.pinw, ch * 2 + 0, lane);
    load_pfrag(b1, sh.pinw, ch * 2 + 1, lane);
    acc0 = wmma_f16(ax, b0, acc0);
    acc1 = wmma_f16(ax, b1, acc1);
  }
  VF xp0, xp1; xp0.v = acc0; xp1.v = acc1;
  {
    float b0 = sh.inb[col], b1 = sh.inb[col + 16];
#pragma unroll
    for (int r = 0; r < 8; ++r) { xp0.f[r] += b0; xp1.f[r] += b1; }
  }

  // ---- recursion state (f32, D-fragment layout), broadcast inits
  VF zH0, zH1, zL0, zL1;
  {
    float h0 = sh.hin[col], h1 = sh.hin[col + 16];
    float l0 = sh.lin[col], l1 = sh.lin[col + 16];
#pragma unroll
    for (int r = 0; r < 8; ++r) {
      zH0.f[r] = h0; zH1.f[r] = h1;
      zL0.f[r] = l0; zL1.f[r] = l1;
    }
  }

  const float g0 = sh.nrm[col], g1 = sh.nrm[col + 16];
  const float ow00 = sh.ow[col][0], ow01 = sh.ow[col + 16][0];
  const float ow10 = sh.ow[col][1], ow11 = sh.ow[col + 16][1];

  for (int hc = 0; hc < 4; ++hc) {
    for (int lc = 0; lc < 3; ++lc) {
      VF t0, t1;
#pragma unroll
      for (int r = 0; r < 8; ++r) {
        t0.f[r] = zL0.f[r] + zH0.f[r] + xp0.f[r];
        t1.f[r] = zL1.f[r] + zH1.f[r] + xp1.f[r];
      }
      block16(t0, t1, sh.zz[wv], sh.act[wv], bw1f, bw2f, bw3f, g0, g1, lane);
      zL0 = t0; zL1 = t1;
    }
    {
      VF t0, t1;
#pragma unroll
      for (int r = 0; r < 8; ++r) {
        t0.f[r] = zH0.f[r] + zL0.f[r];
        t1.f[r] = zH1.f[r] + zL1.f[r];
      }
      block16(t0, t1, sh.zz[wv], sh.act[wv], bw1f, bw2f, bw3f, g0, g1, lane);
      zH0 = t0; zH1 = t1;
    }
    // logits = zH @ out_w + out_b  (32 -> 2), 16-lane-half reductions
#pragma unroll
    for (int r = 0; r < 8; ++r) {
      float p0 = zH0.f[r] * ow00 + zH1.f[r] * ow01;
      float p1 = zH0.f[r] * ow10 + zH1.f[r] * ow11;
      p0 += __shfl_xor(p0, 1, 32); p1 += __shfl_xor(p1, 1, 32);
      p0 += __shfl_xor(p0, 2, 32); p1 += __shfl_xor(p1, 2, 32);
      p0 += __shfl_xor(p0, 4, 32); p1 += __shfl_xor(p1, 4, 32);
      p0 += __shfl_xor(p0, 8, 32); p1 += __shfl_xor(p1, 8, 32);
      if (col == 0) {
        sh.lg[wv][r + 8 * half][0] = p0 + sh.ob[0];
        sh.lg[wv][r + 8 * half][1] = p1 + sh.ob[1];
      }
    }
    lds_fence();
    {
      float vlog = sh.lg[wv][lane >> 1][lane & 1];
      long long rg = row0 + (lane >> 1);
      out[(long long)NB * 2 + (long long)hc * NB * 2 + rg * 2 + (lane & 1)] = vlog;
      if (hc == 3) out[rg * 2 + (lane & 1)] = vlog;   // final logits
    }
    lds_fence();
  }
}

extern "C" void kernel_launch(void* const* d_in, const int* in_sizes, int n_in,
                              void* d_out, int out_size, void* d_ws, size_t ws_size,
                              hipStream_t stream) {
  const float* x      = (const float*)d_in[0];
  const float* in_w   = (const float*)d_in[1];
  const float* in_b   = (const float*)d_in[2];
  const float* w1     = (const float*)d_in[3];
  const float* w2     = (const float*)d_in[4];
  const float* w3     = (const float*)d_in[5];
  const float* norm_w = (const float*)d_in[6];
  const float* Hin    = (const float*)d_in[7];
  const float* Lin    = (const float*)d_in[8];
  const float* out_w  = (const float*)d_in[9];
  const float* out_b  = (const float*)d_in[10];
  // H_cycles / L_cycles (d_in[11], d_in[12]) are fixed constants (4, 3)

  float* out = (float*)d_out;
  const int rows_per_block = NWV * 16;     // 4 waves x 16-row WMMA tiles
  trm_fused<<<NB / rows_per_block, 128, 0, stream>>>(
      x, in_w, in_b, w1, w2, w3, norm_w, Hin, Lin, out_w, out_b, out);
}
